// Block_35648228557144
// MI455X (gfx1250) — matrix operand order repaired and checked by reading, not compiled, not run on hardware
//
#include <hip/hip_runtime.h>
#include <hip/hip_bf16.h>

// ---------------------------------------------------------------------------
// CDNA5 (gfx1250) types / helpers.  wave32; V_WMMA_F32_16X16X32_BF16.
// ---------------------------------------------------------------------------
typedef __attribute__((ext_vector_type(16))) __bf16        bf16x16;
typedef __attribute__((ext_vector_type(8)))  __bf16        bf16x8;
typedef __attribute__((ext_vector_type(8)))  float         f32x8;
typedef __attribute__((ext_vector_type(4)))  float         f32x4;
typedef __attribute__((ext_vector_type(4)))  unsigned int  u32x4;
typedef __attribute__((ext_vector_type(8)))  unsigned int  u32x8;

#define WMMA_BF16(a, b, c) \
  __builtin_amdgcn_wmma_f32_16x16x32_bf16(false, (a), false, (b), (short)0, (c), false, false)

// Two 16B loads -> one 16-element fragment (8 bf16 at p, 8 bf16 at p + 16).
static __device__ __forceinline__ bf16x16 frag_contig(const __bf16* p) {
  bf16x8 lo = *(const bf16x8*)p;
  bf16x8 hi = *(const bf16x8*)(p + 16);
  return __builtin_shufflevector(lo, hi, 0,1,2,3,4,5,6,7,8,9,10,11,12,13,14,15);
}

// A fragment from a row-major 16x32 bf16 tile (ld = 32).
// lanes 0-15: row=lane, K {0..7,16..23}; lanes 16-31: K {8..15,24..31}.
static __device__ __forceinline__ bf16x16 frag_a32(const __bf16* As, int lane) {
  const int row = lane & 15;
  const int khi = (lane & 16) ? 8 : 0;
  bf16x8 lo = *(const bf16x8*)(As + row * 32 + khi);
  bf16x8 hi = *(const bf16x8*)(As + row * 32 + 16 + khi);
  return __builtin_shufflevector(lo, hi, 0,1,2,3,4,5,6,7,8,9,10,11,12,13,14,15);
}

static __device__ __forceinline__ float gelu_exact(float x) {
  return 0.5f * x * (1.0f + erff(x * 0.70710678118654752f));
}

// ---------------------------------------------------------------------------
// Tensor Data Mover: 2-D tile (tile_w x tile_h) of bf16, row stride
// `stride_elems`, HBM -> LDS at byte offset lds_off.  D# per ISA 08 §8.3/8.4.
// Uniform per block; issued by one wave; EXEC is ignored by TDM.
// ---------------------------------------------------------------------------
static __device__ __forceinline__ void tdm_load_2d(unsigned lds_off, const void* gp,
                                                   unsigned tile_w, unsigned tile_h,
                                                   unsigned stride_elems) {
  const unsigned long long ga = (unsigned long long)(uintptr_t)gp;
  u32x4 g0;
  g0[0] = 1u;                                                  // count=1 user D#
  g0[1] = lds_off;                                             // lds_addr (bytes)
  g0[2] = (unsigned)ga;                                        // global_addr[31:0]
  g0[3] = (unsigned)((ga >> 32) & 0x01FFFFFFull) | (2u << 30); // ga[56:32] | type=2
  u32x8 g1;
  g1[0] = 1u << 16;                       // workgroup_mask=0, data_size=1 (2 bytes)
  g1[1] = (tile_w & 0xFFFFu) << 16;       // tensor_dim0[15:0] = tile_w
  g1[2] = (tile_h & 0xFFFFu) << 16;       // tensor_dim0 hi=0 | tensor_dim1 lo = tile_h
  g1[3] = (tile_w & 0xFFFFu) << 16;       // tensor_dim1 hi=0 | tile_dim0 = tile_w
  g1[4] = tile_h & 0xFFFFu;               // tile_dim1 = tile_h, tile_dim2 = 0
  g1[5] = stride_elems;                   // tensor_dim0_stride[31:0]
  g1[6] = 0u;                             // stride hi / dim1_stride lo
  g1[7] = 0u;
  asm volatile("tensor_load_to_lds %0, %1" :: "s"(g0), "s"(g1) : "memory");
}

// ---------------------------------------------------------------------------
// Elementwise f32 -> bf16 (vectorized x4).
// ---------------------------------------------------------------------------
__global__ __launch_bounds__(256) void cvt_bf16(const float* __restrict__ X,
                                                __bf16* __restrict__ Y, int n4) {
  const int i = blockIdx.x * 256 + threadIdx.x;
  if (i < n4) {
    f32x4 v = *(const f32x4*)(X + (size_t)i * 4);
    __bf16* y = Y + (size_t)i * 4;
    y[0] = (__bf16)v.x; y[1] = (__bf16)v.y; y[2] = (__bf16)v.z; y[3] = (__bf16)v.w;
  }
}

// ---------------------------------------------------------------------------
// W[K,N] f32 -> WT[N,K] bf16, 32x32 LDS tiles (coalesced both sides).
// ---------------------------------------------------------------------------
__global__ __launch_bounds__(256) void transpose_cvt(const float* __restrict__ W,
                                                     __bf16* __restrict__ WT,
                                                     int K, int N) {
  __shared__ float t[32][33];
  const int k0 = blockIdx.x * 32, n0 = blockIdx.y * 32;
  const int tx = threadIdx.x & 31, ty = threadIdx.x >> 5;   // 8 rows per step
#pragma unroll
  for (int r = ty; r < 32; r += 8) t[r][tx] = W[(size_t)(k0 + r) * N + n0 + tx];
  __syncthreads();
#pragma unroll
  for (int r = ty; r < 32; r += 8)
    WT[(size_t)(n0 + r) * K + k0 + tx] = (__bf16)t[tx][r];
}

// ---------------------------------------------------------------------------
// Build V^T per (b,h): vt[bh][d][s] = qkvb[b][s][2048 + h*64 + d]  (bf16).
// ---------------------------------------------------------------------------
__global__ __launch_bounds__(256) void vt_build(const __bf16* __restrict__ qkvb,
                                                __bf16* __restrict__ vt) {
  const int s   = blockIdx.x * 256 + threadIdx.x;
  const int bhd = blockIdx.y;               // bh*64 + d
  const int bh = bhd >> 6, d = bhd & 63;
  const int b = bh >> 4, h = bh & 15;
  vt[(size_t)bhd * 2048 + s] =
      qkvb[(size_t)b * 2048 * 3072 + (size_t)s * 3072 + 2048 + h * 64 + d];
}

// ---------------------------------------------------------------------------
// GEMM: C[M,N] = act(A[M,K] @ Bt[N,K]^T + bias[N]).  bf16 in, f32 acc.
// Block 256 (8 waves), tile M=32 x N=128, K-step 32.  LDS tiles are DOUBLE-
// BUFFERED through the Tensor Data Mover: wave 0 issues tile k+1 while all
// waves compute on tile k; s_wait_tensorcnt(2) retires tile k (TDM ops are
// in-order per wave, 2 descriptors per tile).  Fragments: ds_load_b128 pairs.
// ---------------------------------------------------------------------------
template <int ACT, int OUT_BF16>
__global__ __launch_bounds__(256) void gemm_bf16(const __bf16* __restrict__ A,
                                                 const __bf16* __restrict__ Bt,
                                                 const float* __restrict__ bias,
                                                 void* __restrict__ Cout,
                                                 int M, int N, int K) {
  __shared__ __attribute__((aligned(16))) __bf16 As[2][32 * 32];    // [32 M][32 K]
  __shared__ __attribute__((aligned(16))) __bf16 Bs[2][128 * 32];   // [128 N][32 K]
  const int tid = threadIdx.x, wave = tid >> 5, lane = tid & 31;
  const int m0 = blockIdx.x * 32, n0 = blockIdx.y * 128;
  const unsigned lds_a0 = (unsigned)(uintptr_t)&As[0][0];
  const unsigned lds_a1 = (unsigned)(uintptr_t)&As[1][0];
  const unsigned lds_b0 = (unsigned)(uintptr_t)&Bs[0][0];
  const unsigned lds_b1 = (unsigned)(uintptr_t)&Bs[1][0];

  // prologue: tile 0 into buffer 0
  if (wave == 0) {
    tdm_load_2d(lds_a0, A + (size_t)m0 * K, 32, 32, (unsigned)K);
    tdm_load_2d(lds_b0, Bt + (size_t)n0 * K, 32, 128, (unsigned)K);
  }

  f32x8 acc0 = {}, acc1 = {};
  for (int k0 = 0; k0 < K; k0 += 32) {
    const int p = (k0 >> 5) & 1;
    if (wave == 0) {
      if (k0 + 32 < K) {   // issue tile k+1 into the other buffer (reads of it
                           // finished at the previous iteration's barrier)
        tdm_load_2d(p ? lds_a0 : lds_a1, A + (size_t)m0 * K + k0 + 32, 32, 32, (unsigned)K);
        tdm_load_2d(p ? lds_b0 : lds_b1, Bt + (size_t)n0 * K + k0 + 32, 32, 128, (unsigned)K);
        __builtin_amdgcn_s_wait_tensorcnt(2);   // tile k done, tile k+1 in flight
      } else {
        __builtin_amdgcn_s_wait_tensorcnt(0);
      }
    } else if (wave == 7 && lane == 0 && k0 + 64 < K) {
      __builtin_prefetch(A + (size_t)m0 * K + k0 + 64, 0, 1);
      __builtin_prefetch(Bt + (size_t)n0 * K + k0 + 64, 0, 1);
    }
    __syncthreads();                            // tile k visible to all waves
    const __bf16* Ap = As[p];
    const __bf16* Bp = Bs[p];
    const bf16x16 a0 = frag_a32(Ap, lane);
    const bf16x16 a1 = frag_a32(Ap + 16 * 32, lane);
    // B fragment: lane holds col nl; lanes 0-15 K {0..7,16..23}, lanes 16-31 K {8..15,24..31}.
    const int nl = wave * 16 + (lane & 15);
    const int kb = (lane & 16) ? 8 : 0;
    const bf16x16 bb = frag_contig(Bp + nl * 32 + kb);
    acc0 = WMMA_BF16(a0, bb, acc0);
    acc1 = WMMA_BF16(a1, bb, acc1);
    __syncthreads();                            // reads done before buffer reuse
  }

  const int col = n0 + wave * 16 + (lane & 15);
  const int rb  = (lane & 16) ? 8 : 0;
  const float bv = bias[col];
#pragma unroll
  for (int j = 0; j < 8; ++j) {
    float v0 = acc0[j] + bv;
    float v1 = acc1[j] + bv;
    if (ACT) { v0 = gelu_exact(v0); v1 = gelu_exact(v1); }
    const size_t i0 = (size_t)(m0 + rb + j) * N + col;
    const size_t i1 = (size_t)(m0 + 16 + rb + j) * N + col;
    if (OUT_BF16) {
      ((__bf16*)Cout)[i0] = (__bf16)v0;
      ((__bf16*)Cout)[i1] = (__bf16)v1;
    } else {
      ((float*)Cout)[i0] = v0;
      ((float*)Cout)[i1] = v1;
    }
  }
}

// ---------------------------------------------------------------------------
// Flash attention.  qkvb: [B,S,3D] bf16; vt: [B*H, 64, S] bf16; O: [B,S,D] bf16.
// Block 128 (4 waves); each wave owns a 16-row q tile, streams 32 keys/iter.
// ---------------------------------------------------------------------------
__global__ __launch_bounds__(128) void attn_flash(const __bf16* __restrict__ qkvb,
                                                  const __bf16* __restrict__ vt,
                                                  __bf16* __restrict__ O) {
  const int S = 2048, D3 = 3072, Dm = 1024;
  const int bh = blockIdx.y, b = bh >> 4, h = bh & 15;
  const int wave = threadIdx.x >> 5, lane = threadIdx.x & 31;
  const int q0 = (blockIdx.x * 4 + wave) * 16;

  const __bf16* base = qkvb + (size_t)b * S * D3;
  const __bf16* Qp = base + h * 64;
  const __bf16* Kp = base + Dm + h * 64;
  const __bf16* Vt = vt + (size_t)bh * 64 * S;

  // Q A-fragments for hd chunks [0,32), [32,64): two b128 loads each.
  bf16x16 qa[2];
  {
    const int row = lane & 15;
    const int khi = (lane & 16) ? 8 : 0;
#pragma unroll
    for (int half = 0; half < 2; ++half) {
      const __bf16* qr = Qp + (size_t)(q0 + row) * D3 + half * 32;
      bf16x8 lo = *(const bf16x8*)(qr + khi);
      bf16x8 hi = *(const bf16x8*)(qr + 16 + khi);
      qa[half] = __builtin_shufflevector(lo, hi, 0,1,2,3,4,5,6,7,8,9,10,11,12,13,14,15);
    }
  }

  f32x8 oacc[4] = {{}, {}, {}, {}};
  float m_r[8], l_r[8];
#pragma unroll
  for (int j = 0; j < 8; ++j) { m_r[j] = -3.0e38f; l_r[j] = 0.f; }

  __shared__ __attribute__((aligned(16))) __bf16 Ps[4][16 * 32];
  __bf16* P = Ps[wave];
  const int rb = (lane & 16) ? 8 : 0;
  const float scale = 0.125f;   // 1/sqrt(64)

  for (int key0 = 0; key0 <= q0 + 15; key0 += 32) {
    // prefetch next key block (K rows + a slice of V^T)
    if (key0 + 32 <= q0 + 15) {
      __builtin_prefetch(Kp + (size_t)(key0 + 32 + (lane & 31)) * D3, 0, 1);
      __builtin_prefetch(Vt + (size_t)(lane * 2) * S + key0 + 32, 0, 1);
    }
    // ---- S = Q K^T
    f32x8 sc[2] = {{}, {}};
#pragma unroll
    for (int gde = 0; gde < 2; ++gde) {
      const int kcol = key0 + gde * 16 + (lane & 15);
      const int hdo  = (lane & 16) ? 8 : 0;
#pragma unroll
      for (int half = 0; half < 2; ++half) {
        const bf16x16 bb = frag_contig(Kp + (size_t)kcol * D3 + half * 32 + hdo);
        sc[gde] = WMMA_BF16(qa[half], bb, sc[gde]);
      }
    }

    // ---- causal mask + online softmax (row reductions in 16-lane halves)
#pragma unroll
    for (int j = 0; j < 8; ++j) {
      const int qi = q0 + rb + j;
      float s0 = sc[0][j] * scale; if (key0 + (lane & 15) > qi)      s0 = -10000.f;
      float s1 = sc[1][j] * scale; if (key0 + 16 + (lane & 15) > qi) s1 = -10000.f;
      float mx = fmaxf(s0, s1);
#pragma unroll
      for (int off = 8; off > 0; off >>= 1) mx = fmaxf(mx, __shfl_xor(mx, off, 16));
      const float mnew  = fmaxf(m_r[j], mx);
      const float alpha = __expf(m_r[j] - mnew);
      const float p0 = __expf(s0 - mnew);
      const float p1 = __expf(s1 - mnew);
      float psum = p0 + p1;
#pragma unroll
      for (int off = 8; off > 0; off >>= 1) psum += __shfl_xor(psum, off, 16);
      l_r[j] = l_r[j] * alpha + psum;
      m_r[j] = mnew;
#pragma unroll
      for (int dc = 0; dc < 4; ++dc) oacc[dc][j] *= alpha;
      P[(rb + j) * 32 + (lane & 15)]      = (__bf16)p0;
      P[(rb + j) * 32 + 16 + (lane & 15)] = (__bf16)p1;
    }
    asm volatile("s_wait_dscnt 0" ::: "memory");

    // ---- O += P @ V   (V^T rows are contiguous in s: b128 fragment loads)
    const bf16x16 pa = frag_a32(P, lane);
    const int kb = (lane & 16) ? 8 : 0;
#pragma unroll
    for (int dc = 0; dc < 4; ++dc) {
      const int d = dc * 16 + (lane & 15);
      const bf16x16 vb = frag_contig(Vt + (size_t)d * S + key0 + kb);
      oacc[dc] = WMMA_BF16(pa, vb, oacc[dc]);
    }
  }

  __bf16* orow = O + (size_t)b * S * Dm + h * 64;
#pragma unroll
  for (int j = 0; j < 8; ++j) {
    const float inv = 1.0f / l_r[j];
#pragma unroll
    for (int dc = 0; dc < 4; ++dc)
      orow[(size_t)(q0 + rb + j) * Dm + dc * 16 + (lane & 15)] = (__bf16)(oacc[dc][j] * inv);
  }
}

// ---------------------------------------------------------------------------
// out = layernorm(A + B) * g + be; optionally also emit bf16 copy.  D=1024.
// ---------------------------------------------------------------------------
template <int WRITE_B>
__global__ __launch_bounds__(256) void add_ln(const float* __restrict__ A,
                                              const float* __restrict__ Bv,
                                              const float* __restrict__ g,
                                              const float* __restrict__ be,
                                              float* __restrict__ out,
                                              __bf16* __restrict__ outb) {
  const int D = 1024;
  const int row = blockIdx.x, tid = threadIdx.x;
  const float* pa = A + (size_t)row * D;
  const float* pb = Bv + (size_t)row * D;
  float v[4]; float s = 0.f, ss = 0.f;
#pragma unroll
  for (int i = 0; i < 4; ++i) {
    const int c = tid + i * 256;
    const float t = pa[c] + pb[c];
    v[i] = t; s += t; ss += t * t;
  }
#pragma unroll
  for (int off = 16; off > 0; off >>= 1) { s += __shfl_xor(s, off, 32); ss += __shfl_xor(ss, off, 32); }
  __shared__ float shs[8], shq[8];
  if ((tid & 31) == 0) { shs[tid >> 5] = s; shq[tid >> 5] = ss; }
  __syncthreads();
  s = 0.f; ss = 0.f;
#pragma unroll
  for (int w = 0; w < 8; ++w) { s += shs[w]; ss += shq[w]; }
  const float mu  = s * (1.0f / D);
  const float var = ss * (1.0f / D) - mu * mu;
  const float r   = rsqrtf(var + 1e-5f);
#pragma unroll
  for (int i = 0; i < 4; ++i) {
    const int c = tid + i * 256;
    const float o = (v[i] - mu) * r * g[c] + be[c];
    out[(size_t)row * D + c] = o;
    if (WRITE_B) outb[(size_t)row * D + c] = (__bf16)o;
  }
}

// ---------------------------------------------------------------------------
// Driver.  B=2, S=2048, D=1024, F=4096 -> M = 4096 token rows.
// ---------------------------------------------------------------------------
extern "C" void kernel_launch(void* const* d_in, const int* in_sizes, int n_in,
                              void* d_out, int out_size, void* d_ws, size_t ws_size,
                              hipStream_t stream) {
  const float* x      = (const float*)d_in[0];
  const float* w_attn = (const float*)d_in[1];
  const float* b_attn = (const float*)d_in[2];
  const float* w_o    = (const float*)d_in[3];
  const float* b_o    = (const float*)d_in[4];
  const float* ln1_g  = (const float*)d_in[5];
  const float* ln1_b  = (const float*)d_in[6];
  const float* w_fc   = (const float*)d_in[7];
  const float* b_fc   = (const float*)d_in[8];
  const float* w_pr   = (const float*)d_in[9];
  const float* b_pr   = (const float*)d_in[10];
  const float* ln2_g  = (const float*)d_in[11];
  const float* ln2_b  = (const float*)d_in[12];
  float* out = (float*)d_out;

  const size_t M = 4096;
  char* p = (char*)d_ws;
  auto alloc = [&](size_t bytes) { char* r = p; p += (bytes + 255) & ~(size_t)255; return r; };
  __bf16* xb    = (__bf16*)alloc(M * 1024 * 2);
  __bf16* wAT   = (__bf16*)alloc((size_t)3072 * 1024 * 2);   // w_attn^T [3072,1024]
  __bf16* wOT   = (__bf16*)alloc((size_t)1024 * 1024 * 2);
  __bf16* wFT   = (__bf16*)alloc((size_t)4096 * 1024 * 2);
  __bf16* wPT   = (__bf16*)alloc((size_t)1024 * 4096 * 2);
  __bf16* qkvb  = (__bf16*)alloc(M * 3072 * 2);
  __bf16* vtb   = (__bf16*)alloc((size_t)32 * 64 * 2048 * 2);
  __bf16* obuf  = (__bf16*)alloc(M * 1024 * 2);
  __bf16* n1b   = (__bf16*)alloc(M * 1024 * 2);
  __bf16* fcb   = (__bf16*)alloc(M * 4096 * 2);
  float*  aproj = (float*)alloc(M * 1024 * 4);
  float*  n1f   = (float*)alloc(M * 1024 * 4);
  float*  mbuf  = (float*)alloc(M * 1024 * 4);

  const dim3 blk(256);
  // one-time precision/layout prep (per call; cheap vs 23.3 TB/s)
  cvt_bf16<<<dim3(4096), blk, 0, stream>>>(x, xb, 4096 * 256);
  transpose_cvt<<<dim3(32, 96),  blk, 0, stream>>>(w_attn, wAT, 1024, 3072);
  transpose_cvt<<<dim3(32, 32),  blk, 0, stream>>>(w_o,    wOT, 1024, 1024);
  transpose_cvt<<<dim3(32, 128), blk, 0, stream>>>(w_fc,   wFT, 1024, 4096);
  transpose_cvt<<<dim3(128, 32), blk, 0, stream>>>(w_pr,   wPT, 4096, 1024);

  // qkv = x @ w_attn + b_attn                    (bf16 out)
  gemm_bf16<0, 1><<<dim3(128, 24), blk, 0, stream>>>(xb, wAT, b_attn, qkvb, 4096, 3072, 1024);
  // V^T layout for the PV matmul
  vt_build<<<dim3(8, 2048), blk, 0, stream>>>(qkvb, vtb);
  // causal flash attention
  attn_flash<<<dim3(32, 32), dim3(128), 0, stream>>>(qkvb, vtb, obuf);
  // attn out projection                          (f32 out, residual carrier)
  gemm_bf16<0, 0><<<dim3(128, 8), blk, 0, stream>>>(obuf, wOT, b_o, aproj, 4096, 1024, 1024);
  // n1 = LN(x + attn)                            (f32 + bf16 copies)
  add_ln<1><<<4096, blk, 0, stream>>>(x, aproj, ln1_g, ln1_b, n1f, n1b);
  // fc = gelu(n1 @ w_fc + b_fc)                  (bf16 out)
  gemm_bf16<1, 1><<<dim3(128, 32), blk, 0, stream>>>(n1b, wFT, b_fc, fcb, 4096, 4096, 1024);
  // mlp = fc @ w_pr + b_pr                       (f32 out)
  gemm_bf16<0, 0><<<dim3(128, 8), blk, 0, stream>>>(fcb, wPT, b_pr, mbuf, 4096, 1024, 4096);
  // out = LN(n1 + mlp)
  add_ln<0><<<4096, blk, 0, stream>>>(n1f, mbuf, ln2_g, ln2_b, out, nullptr);
}
